// SCFT_15152644621141
// MI455X (gfx1250) — matrix operand order repaired and checked by reading, not compiled, not run on hardware
//
#include <hip/hip_runtime.h>
#include <hip/hip_bf16.h>
#include <math.h>

typedef _Float16 half_t;
typedef __attribute__((ext_vector_type(16))) _Float16 v16h;
typedef __attribute__((ext_vector_type(8)))  _Float16 v8h;
typedef __attribute__((ext_vector_type(4)))  _Float16 v4h;
typedef __attribute__((ext_vector_type(8)))  float    v8f;
typedef __attribute__((ext_vector_type(4)))  float    v4f;

static constexpr int BSc  = 16;
static constexpr int Dc   = 448;   // 7 * 64
static constexpr int HWc  = 4096;  // 64 * 64

// ---------------------------------------------------------------------------
// helpers: CDNA5 async global->LDS copy (16B per lane), ASYNCcnt tracked
// ---------------------------------------------------------------------------
__device__ __forceinline__ void async_copy_b128(uint32_t lds_addr, uint64_t gaddr) {
    asm volatile("global_load_async_to_lds_b128 %0, %1, off"
                 :: "v"(lds_addr), "v"(gaddr) : "memory");
}
__device__ __forceinline__ void wait_async_le4() {
    asm volatile("s_wait_asynccnt 0x4" ::: "memory");
}
__device__ __forceinline__ void wait_async_0() {
    asm volatile("s_wait_asynccnt 0x0" ::: "memory");
}

// ---------------------------------------------------------------------------
// f32 -> f16 conversion (weights)
// ---------------------------------------------------------------------------
__global__ void cvt_f32_f16(const float* __restrict__ src, half_t* __restrict__ dst, int n) {
    int i = blockIdx.x * blockDim.x + threadIdx.x;
    if (i < n) dst[i] = (half_t)src[i];
}

// ---------------------------------------------------------------------------
// Tiled transpose + convert:  in (b, C, H) f32  ->  out (b, H, C) f16
// 32x32 tiles through LDS; coalesced on both sides.
// ---------------------------------------------------------------------------
__global__ __launch_bounds__(256)
void transpose_cvt(const float* __restrict__ in, half_t* __restrict__ out, int C, int H)
{
    __shared__ float tile[32][33];
    const int b  = blockIdx.z;
    const int c0 = blockIdx.y * 32;
    const int h0 = blockIdx.x * 32;
    const int tr = threadIdx.x >> 3;        // 0..31
    const int tc = (threadIdx.x & 7) * 4;   // 0,4,..,28

    const float* pin = in + (long)b * C * H + (long)(c0 + tr) * H + h0 + tc;
    v4f v = *(const v4f*)pin;
    tile[tr][tc + 0] = v[0]; tile[tr][tc + 1] = v[1];
    tile[tr][tc + 2] = v[2]; tile[tr][tc + 3] = v[3];
    __syncthreads();

    v4h o;
    #pragma unroll
    for (int i = 0; i < 4; ++i) o[i] = (half_t)tile[tc + i][tr];
    half_t* pout = out + (long)b * C * H + (long)(h0 + tr) * C + c0 + tc;
    *(v4h*)pout = o;
}

// ---------------------------------------------------------------------------
// 64x64-tile WMMA GEMM, single fast path:
//   C[b] = scale * A[b] (MxK f16 row-major)  x  B[b]^T   (B stored NxK f16)
// 128 threads = 4 waves; each wave owns a 32x32 micro-tile = 4x
// v_wmma_f32_16x16x32_f16 per K-step.  Global->LDS staging is fully
// asynchronous (global_load_async_to_lds_b128) with LDS double buffering.
//   C_TRANS : store C transposed (C[n*M + m]) -- used to produce Vv^T
//   LTR     : fused epilogue sum(relu(acc - v_asta + 12)) -> atomicAdd
// ---------------------------------------------------------------------------
template<bool C_TRANS, bool LTR, typename OutT>
__global__ __launch_bounds__(128)
void gemm64x64_wmma(const half_t* __restrict__ A,
                    const half_t* __restrict__ B,
                    OutT*         __restrict__ C,
                    const float*  __restrict__ Vasta,
                    float*        __restrict__ gScalar,
                    int M, int N, int K,
                    long aBatchStride, long bBatchStride, long cBatchStride,
                    int nBatch, int bRoll, float scale, float invTotal)
{
    constexpr int LDSS = 40;                    // padded row stride (80B)
    __shared__ half_t ldsA[2][64 * LDSS];       // [m][k]
    __shared__ half_t ldsB[2][64 * LDSS];       // [n][k]

    const int t     = threadIdx.x;
    const int lane  = t & 31;
    const int wave  = t >> 5;
    const int waveM = wave & 1;
    const int waveN = wave >> 1;
    const int l16   = lane & 15;
    const int lHi   = lane >> 4;

    const int b      = blockIdx.z;
    const int mBlock = blockIdx.y * 64;
    const int nBlock = blockIdx.x * 64;

    const half_t* Ab = A + (long)b * aBatchStride;
    const half_t* Bb = B + (long)((b + bRoll) % nBatch) * bBatchStride;

    // per-thread staging coordinates: 2 chunks of 8 halves (16B) each,
    // identical indexing for A[m][k] and B[n][k] tiles (64 rows x 32 k).
    auto issueStage = [&](int buf, int kk) {
        #pragma unroll
        for (int c = 0; c < 2; ++c) {
            int linear = t * 16 + c * 8;
            int row = linear >> 5, col = linear & 31;
            uint32_t la = (uint32_t)(uintptr_t)&ldsA[buf][row * LDSS + col];
            uint64_t ga = (uint64_t)(uintptr_t)(Ab + (long)(mBlock + row) * K + kk + col);
            async_copy_b128(la, ga);
            uint32_t lb = (uint32_t)(uintptr_t)&ldsB[buf][row * LDSS + col];
            uint64_t gb = (uint64_t)(uintptr_t)(Bb + (long)(nBlock + row) * K + kk + col);
            async_copy_b128(lb, gb);
        }
    };

    v8f acc[2][2] = {};

    const int nIter = K >> 5;
    issueStage(0, 0);
    for (int it = 0; it < nIter; ++it) {
        const int buf  = it & 1;
        const bool more = (it + 1) < nIter;
        if (more) { issueStage(buf ^ 1, (it + 1) << 5); wait_async_le4(); }
        else      { wait_async_0(); }
        __syncthreads();                       // all waves' async data visible

        v16h aFrag[2], bFrag[2];
        #pragma unroll
        for (int mi = 0; mi < 2; ++mi) {
            // 16-bit A layout: lanes 0-15 hold K {0..7,16..23}; lanes 16-31 {8..15,24..31}
            const half_t* p = &ldsA[buf][(waveM * 32 + mi * 16 + l16) * LDSS];
            v8h lo = *(const v8h*)(p + lHi * 8);
            v8h hi = *(const v8h*)(p + 16 + lHi * 8);
            #pragma unroll
            for (int i = 0; i < 8; ++i) { aFrag[mi][i] = lo[i]; aFrag[mi][8 + i] = hi[i]; }
        }
        #pragma unroll
        for (int ni = 0; ni < 2; ++ni) {
            // 16-bit B layout: lanes 0-15 hold K {0..7,16..23}; lanes 16-31 {8..15,24..31} (col = lane%16)
            const half_t* p = &ldsB[buf][(waveN * 32 + ni * 16 + l16) * LDSS];
            v8h lo = *(const v8h*)(p + lHi * 8);
            v8h hi = *(const v8h*)(p + 16 + lHi * 8);
            #pragma unroll
            for (int i = 0; i < 8; ++i) { bFrag[ni][i] = lo[i]; bFrag[ni][8 + i] = hi[i]; }
        }

        #pragma unroll
        for (int mi = 0; mi < 2; ++mi)
            #pragma unroll
            for (int ni = 0; ni < 2; ++ni)
                acc[mi][ni] = __builtin_amdgcn_wmma_f32_16x16x32_f16(
                    false, aFrag[mi], false, bFrag[ni],
                    (short)0, acc[mi][ni], false, false);
        __syncthreads();                       // reads done before buffer reuse
    }

    // ---- epilogue: C layout VGPR r, lanes 0-15 -> M=r, lanes 16-31 -> M=8+r; N=lane%16
    if constexpr (LTR) {
        const float* Vb = Vasta + (long)b * cBatchStride;
        float lsum = 0.f;
        #pragma unroll
        for (int mi = 0; mi < 2; ++mi)
            #pragma unroll
            for (int ni = 0; ni < 2; ++ni)
                #pragma unroll
                for (int r = 0; r < 8; ++r) {
                    int m = mBlock + waveM * 32 + mi * 16 + lHi * 8 + r;
                    int n = nBlock + waveN * 32 + ni * 16 + l16;
                    float vneg = acc[mi][ni][r] * scale;
                    lsum += fmaxf(vneg - Vb[(long)m * N + n] + 12.0f, 0.0f);
                }
        #pragma unroll
        for (int off = 16; off > 0; off >>= 1) lsum += __shfl_xor(lsum, off, 32);
        if (lane == 0) atomicAdd(gScalar, lsum * invTotal);
    } else {
        OutT* Cb = C + (long)b * cBatchStride;
        #pragma unroll
        for (int mi = 0; mi < 2; ++mi)
            #pragma unroll
            for (int ni = 0; ni < 2; ++ni)
                #pragma unroll
                for (int r = 0; r < 8; ++r) {
                    int m = mBlock + waveM * 32 + mi * 16 + lHi * 8 + r;
                    int n = nBlock + waveN * 32 + ni * 16 + l16;
                    float v = acc[mi][ni][r] * scale;
                    if constexpr (C_TRANS) Cb[(long)n * M + m] = (OutT)v;
                    else                   Cb[(long)m * N + n] = (OutT)v;
                }
    }
}

// ---------------------------------------------------------------------------
// Row softmax: one wave per row of length 448, f32 in -> f16 out.
// ---------------------------------------------------------------------------
__global__ __launch_bounds__(256)
void softmax_rows(const float* __restrict__ S, half_t* __restrict__ Out,
                  int rowLen, int nRows)
{
    int wave = threadIdx.x >> 5;
    int lane = threadIdx.x & 31;
    int row  = blockIdx.x * 8 + wave;
    if (row >= nRows) return;
    const float* p = S + (long)row * rowLen;

    float mx = -3.0e38f;
    for (int i = lane; i < rowLen; i += 32) mx = fmaxf(mx, p[i]);
    #pragma unroll
    for (int off = 16; off > 0; off >>= 1) mx = fmaxf(mx, __shfl_xor(mx, off, 32));

    float sum = 0.f;
    for (int i = lane; i < rowLen; i += 32) sum += __expf(p[i] - mx);
    #pragma unroll
    for (int off = 16; off > 0; off >>= 1) sum += __shfl_xor(sum, off, 32);

    float inv = 1.0f / sum;
    half_t* q = Out + (long)row * rowLen;
    for (int i = lane; i < rowLen; i += 32) q[i] = (half_t)(__expf(p[i] - mx) * inv);
}

// ---------------------------------------------------------------------------
__global__ void add_residual(const float* __restrict__ Vs, const float* __restrict__ Va,
                             float* __restrict__ out, long n4)
{
    long i      = (long)blockIdx.x * blockDim.x + threadIdx.x;
    long stride = (long)gridDim.x * blockDim.x;
    const v4f* a = (const v4f*)Vs;
    const v4f* b = (const v4f*)Va;
    v4f* o = (v4f*)out;
    for (; i < n4; i += stride) o[i] = a[i] + b[i];
}

__global__ void zero_scalar(float* p) { if (threadIdx.x == 0 && blockIdx.x == 0) *p = 0.f; }

// ---------------------------------------------------------------------------
extern "C" void kernel_launch(void* const* d_in, const int* in_sizes, int n_in,
                              void* d_out, int out_size, void* d_ws, size_t ws_size,
                              hipStream_t stream)
{
    const float* V_r = (const float*)d_in[0];
    const float* V_s = (const float*)d_in[1];
    const float* W_q = (const float*)d_in[2];
    const float* W_k = (const float*)d_in[3];
    const float* W_v = (const float*)d_in[4];

    const long bigN   = (long)BSc * Dc * HWc;       // 29,360,128
    const long scoreN = (long)BSc * Dc * Dc;        // 3,211,264

    // ---- workspace carve-out (256B aligned) ----
    char* w = (char*)d_ws;
    auto carve = [&](size_t bytes) { char* p = w; w += (bytes + 255) & ~(size_t)255; return p; };
    half_t* Wq16     = (half_t*)carve((size_t)Dc * Dc * 2);
    half_t* Wk16     = (half_t*)carve((size_t)Dc * Dc * 2);
    half_t* Wv16     = (half_t*)carve((size_t)Dc * Dc * 2);
    half_t* Vst16    = (half_t*)carve((size_t)bigN * 2);    // V_s^T  (b,h,c) f16
    half_t* Vrt16    = (half_t*)carve((size_t)bigN * 2);    // V_r^T  (b,h,c) f16
    half_t* Q16      = (half_t*)carve((size_t)bigN * 2);    // wq_vs  (b,d,h)
    half_t* K16      = (half_t*)carve((size_t)bigN * 2);    // wk_vr  (b,e,h)  (NxK for scores)
    half_t* Vv16t    = (half_t*)carve((size_t)bigN * 2);    // wv_vr^T (b,h,e) (NxK for apply)
    float*  scoresF  = (float*) carve((size_t)scoreN * 4);
    half_t* alpha16  = (half_t*)carve((size_t)scoreN * 2);
    half_t* alphaN16 = (half_t*)carve((size_t)scoreN * 2);
    float*  vasta    = (float*) carve((size_t)bigN * 4);

    float* outC = (float*)d_out;
    float* outL = outC + bigN;

    const float invSqrtD = 1.0f / sqrtf((float)Dc);
    const float invTotal = 1.0f / (float)bigN;

    dim3 blk(128);

    // 1) weight conversion
    cvt_f32_f16<<<(Dc * Dc + 255) / 256, 256, 0, stream>>>(W_q, Wq16, Dc * Dc);
    cvt_f32_f16<<<(Dc * Dc + 255) / 256, 256, 0, stream>>>(W_k, Wk16, Dc * Dc);
    cvt_f32_f16<<<(Dc * Dc + 255) / 256, 256, 0, stream>>>(W_v, Wv16, Dc * Dc);

    // 2) transpose+convert inputs once: (b,c,h) f32 -> (b,h,c) f16
    dim3 gT(HWc / 32, Dc / 32, BSc);
    transpose_cvt<<<gT, 256, 0, stream>>>(V_s, Vst16, Dc, HWc);
    transpose_cvt<<<gT, 256, 0, stream>>>(V_r, Vrt16, Dc, HWc);

    // 3) projections: out[d,h] = sum_c W[d,c] * Vt[h,c]   (B is NxK)
    dim3 gProj(HWc / 64, Dc / 64, BSc);
    gemm64x64_wmma<false, false, half_t><<<gProj, blk, 0, stream>>>(
        Wq16, Vst16, Q16, nullptr, nullptr, Dc, HWc, Dc,
        0L, (long)Dc * HWc, (long)Dc * HWc, BSc, 0, 1.0f, 0.f);
    gemm64x64_wmma<false, false, half_t><<<gProj, blk, 0, stream>>>(
        Wk16, Vrt16, K16, nullptr, nullptr, Dc, HWc, Dc,
        0L, (long)Dc * HWc, (long)Dc * HWc, BSc, 0, 1.0f, 0.f);
    gemm64x64_wmma<true, false, half_t><<<gProj, blk, 0, stream>>>(   // store transposed
        Wv16, Vrt16, Vv16t, nullptr, nullptr, Dc, HWc, Dc,
        0L, (long)Dc * HWc, (long)Dc * HWc, BSc, 0, 1.0f, 0.f);

    // 4) scores = Q x K^T / sqrt(d)
    dim3 gScore(Dc / 64, Dc / 64, BSc);
    gemm64x64_wmma<false, false, float><<<gScore, blk, 0, stream>>>(
        Q16, K16, scoresF, nullptr, nullptr, Dc, Dc, HWc,
        (long)Dc * HWc, (long)Dc * HWc, (long)Dc * Dc, BSc, 0, invSqrtD, 0.f);
    softmax_rows<<<(BSc * Dc) / 8, 256, 0, stream>>>(scoresF, alpha16, Dc, BSc * Dc);

    // 5) negative scores: keys rolled by one batch
    gemm64x64_wmma<false, false, float><<<gScore, blk, 0, stream>>>(
        Q16, K16, scoresF, nullptr, nullptr, Dc, Dc, HWc,
        (long)Dc * HWc, (long)Dc * HWc, (long)Dc * Dc, BSc, 1, invSqrtD, 0.f);
    softmax_rows<<<(BSc * Dc) / 8, 256, 0, stream>>>(scoresF, alphaN16, Dc, BSc * Dc);

    // 6) v_asta = alpha x wv_vr   (B = Vv^T, NxK)
    gemm64x64_wmma<false, false, float><<<gProj, blk, 0, stream>>>(
        alpha16, Vv16t, vasta, nullptr, nullptr, Dc, HWc, Dc,
        (long)Dc * Dc, (long)Dc * HWc, (long)Dc * HWc, BSc, 0, 1.0f, 0.f);

    // 7) fused negative branch + L_tr reduction (no v_negative buffer)
    zero_scalar<<<1, 1, 0, stream>>>(outL);
    gemm64x64_wmma<false, true, float><<<gProj, blk, 0, stream>>>(
        alphaN16, Vv16t, (float*)nullptr, vasta, outL, Dc, HWc, Dc,
        (long)Dc * Dc, (long)Dc * HWc, (long)Dc * HWc, BSc, 0, 1.0f, invTotal);

    // 8) c_i = V_s + v_asta
    add_residual<<<2048, 256, 0, stream>>>(V_s, vasta, outC, bigN / 4);

    (void)in_sizes; (void)n_in; (void)out_size; (void)ws_size;
}
